// Multihead_self_attention_50397146251425
// MI455X (gfx1250) — matrix operand order repaired and checked by reading, not compiled, not run on hardware
//
#include <hip/hip_runtime.h>

// ---------------------------------------------------------------------------
// Multihead self-attention with RoPE for MI455X (gfx1250, wave32, WMMA).
// All matmuls run on v_wmma_f32_16x16x32_bf16 (bf16 in, f32 accumulate).
// GEMM kernels compute a 16x64 tile per wave: one A fragment feeds 4 WMMAs
// (2.5 b128 loads per WMMA instead of 4), keeping the matrix pipes fed.
// ---------------------------------------------------------------------------

typedef __attribute__((ext_vector_type(16))) __bf16 bf16x16;
typedef __attribute__((ext_vector_type(8)))  float  f32x8;

namespace {
constexpr int Bn   = 4;
constexpr int Sn   = 2048;
constexpr int Dm   = 1024;
constexpr int Hn   = 16;
constexpr int Dk   = 64;
constexpr int Rows = Bn * Sn;          // 8192 token rows
}

#define DEVFN __device__ __forceinline__

// fp32 -> bf16 (round to nearest even)
DEVFN unsigned short f2bf(float f) {
  unsigned u = __float_as_uint(f);
  unsigned r = u + 0x7FFFu + ((u >> 16) & 1u);
  return (unsigned short)(r >> 16);
}

// trivially-copyable 128-bit chunk so unions with __bf16 vectors stay POD
struct __align__(16) U128 { unsigned int x, y, z, w; };
union Frag32 { U128 q[2]; bf16x16 v; };

// load a 16-element bf16 fragment as two contiguous 16-byte chunks
DEVFN bf16x16 ld_frag(const unsigned short* p0, const unsigned short* p1) {
  Frag32 f;
  f.q[0] = *reinterpret_cast<const U128*>(p0);
  f.q[1] = *reinterpret_cast<const U128*>(p1);
  return f.v;
}

DEVFN f32x8 wmma_bf16(bf16x16 a, bf16x16 b, f32x8 c) {
  // (neg_a, A, neg_b, B, c_mod, C, reuse_a, reuse_b)
  return __builtin_amdgcn_wmma_f32_16x16x32_bf16(false, a, false, b,
                                                 (short)0, c, false, false);
}

// reductions across the 16 lanes of a wave half (bits 0..3 only)
DEVFN float rowmax16(float x) {
  x = fmaxf(x, __shfl_xor(x, 1, 32));
  x = fmaxf(x, __shfl_xor(x, 2, 32));
  x = fmaxf(x, __shfl_xor(x, 4, 32));
  x = fmaxf(x, __shfl_xor(x, 8, 32));
  return x;
}
DEVFN float rowsum16(float x) {
  x += __shfl_xor(x, 1, 32);
  x += __shfl_xor(x, 2, 32);
  x += __shfl_xor(x, 4, 32);
  x += __shfl_xor(x, 8, 32);
  return x;
}

// ---------------------------------------------------------------------------
// Kernel 0: fp32 -> bf16 conversion (grid-stride)
// ---------------------------------------------------------------------------
__global__ void cvt_f32_bf16(const float* __restrict__ in,
                             unsigned short* __restrict__ out, int n) {
  int i = blockIdx.x * blockDim.x + threadIdx.x;
  int stride = gridDim.x * blockDim.x;
  for (; i < n; i += stride) out[i] = f2bf(in[i]);
}

// ---------------------------------------------------------------------------
// Kernel 1: fused QKV projection (Y = X * W^T) + RoPE epilogue.
// blockIdx.z: 0 = Q (rope), 1 = K (rope), 2 = V (stored transposed).
// One wave computes a 16x64 tile (exactly one head's features); A fragment
// is reused across 4 B fragments per k-step. 4 waves/block -> 256 columns.
// ---------------------------------------------------------------------------
__global__ __launch_bounds__(128)
void qkv_rope_gemm(const unsigned short* __restrict__ xb,
                   const unsigned short* __restrict__ wqb,
                   const unsigned short* __restrict__ wkb,
                   const unsigned short* __restrict__ wvb,
                   const int* __restrict__ tok,
                   unsigned short* __restrict__ Qb,   // [B,H,S,Dk]
                   unsigned short* __restrict__ Kb,   // [B,H,S,Dk]
                   unsigned short* __restrict__ VT) { // [B,H,Dk,S]
  const int lane  = threadIdx.x & 31;
  const int wave  = threadIdx.x >> 5;
  const int hi    = lane >> 4;      // wave half
  const int lo    = lane & 15;
  const int mbase = blockIdx.x * 16;
  const int ebase = (blockIdx.y * 4 + wave) * 64;   // one head per wave
  const int z     = blockIdx.z;
  const unsigned short* wb = (z == 0) ? wqb : (z == 1) ? wkb : wvb;

  // A: x rows (M = lo); half 'hi' holds K = hi*16+e (contiguous, matches B)
  const unsigned short* arow = xb + (size_t)(mbase + lo) * Dm;
  // B: W^T columns come straight from row-major W rows (N = lo, K contiguous)
  const unsigned short* brow0 = wb + (size_t)(ebase + lo) * Dm;
  const unsigned short* brow1 = brow0 + (size_t)16 * Dm;
  const unsigned short* brow2 = brow0 + (size_t)32 * Dm;
  const unsigned short* brow3 = brow0 + (size_t)48 * Dm;

  f32x8 acc[4] = {};
  for (int kb = 0; kb < Dm; kb += 32) {
    bf16x16 a = ld_frag(arow + kb + hi * 16, arow + kb + hi * 16 + 8);
    bf16x16 b0 = ld_frag(brow0 + kb + hi * 16, brow0 + kb + hi * 16 + 8);
    bf16x16 b1 = ld_frag(brow1 + kb + hi * 16, brow1 + kb + hi * 16 + 8);
    bf16x16 b2 = ld_frag(brow2 + kb + hi * 16, brow2 + kb + hi * 16 + 8);
    bf16x16 b3 = ld_frag(brow3 + kb + hi * 16, brow3 + kb + hi * 16 + 8);
    acc[0] = wmma_bf16(a, b0, acc[0]);
    acc[1] = wmma_bf16(a, b1, acc[1]);
    acc[2] = wmma_bf16(a, b2, acc[2]);
    acc[3] = wmma_bf16(a, b3, acc[3]);
  }

  const int h = ebase >> 6;         // wave-uniform head index

  if (z < 2) {
    unsigned short* dst = (z == 0) ? Qb : Kb;
#pragma unroll
    for (int dt = 0; dt < 4; ++dt) {
      const int dk   = dt * 16 + lo;           // feature within head
      const int pair = dk >> 1;
      const float freq =
          __expf(-(float)(2 * pair) * (9.210340371976184f / 64.f));
#pragma unroll
      for (int r = 0; r < 8; ++r) {
        const int m = mbase + r + hi * 8;      // C layout: M = r + 8*hi
        const int s = m & (Sn - 1);
        const int bi = m >> 11;
        float v = acc[dt][r];
        float p = __shfl_xor(v, 1, 32);        // partner feature (dk ^ 1)
        float sn, cs;
        __sincosf((float)tok[s] * freq, &sn, &cs);
        float out = (dk & 1) ? fmaf(p, sn, v * cs)      // odd:  xe*sin+xo*cos
                             : fmaf(v, cs, -(p * sn));  // even: xe*cos-xo*sin
        dst[(((size_t)bi * Hn + h) * Sn + s) * Dk + dk] = f2bf(out);
      }
    }
  } else {
    // V stored transposed [B,H,Dk,S] so the P*V B-operand loads contiguously.
#pragma unroll
    for (int dt = 0; dt < 4; ++dt) {
      const int dk = dt * 16 + lo;
#pragma unroll
      for (int r = 0; r < 8; ++r) {
        const int m = mbase + r + hi * 8;
        const int s = m & (Sn - 1);
        const int bi = m >> 11;
        VT[(((size_t)bi * Hn + h) * Dk + dk) * Sn + s] = f2bf(acc[dt][r]);
      }
    }
  }
}

// ---------------------------------------------------------------------------
// Kernel 2: causal flash attention. One wave per (b, h, 16-row q tile).
// Online softmax over 32-wide key blocks; P transposed via LDS to A-layout.
// ---------------------------------------------------------------------------
__global__ __launch_bounds__(32)
void flash_attn(const unsigned short* __restrict__ Qb,
                const unsigned short* __restrict__ Kb,
                const unsigned short* __restrict__ VT,
                unsigned short* __restrict__ Ob) {  // [B,S,Dm] head-concat
  const int lane  = threadIdx.x & 31;
  const int hi    = lane >> 4;
  const int lo    = lane & 15;
  const int qtile = blockIdx.x;
  const int bh    = blockIdx.y;
  const int b     = bh >> 4;
  const int h     = bh & 15;
  const int q0    = qtile * 16;

  const unsigned short* qbase = Qb + ((size_t)bh * Sn + q0) * Dk;
  const unsigned short* kbase = Kb + (size_t)bh * Sn * Dk;
  const unsigned short* vbase = VT + (size_t)bh * Dk * Sn;

  // Q tile as two A-fragments (d = 0..31 and 32..63), resident in VGPRs
  const unsigned short* qrow = qbase + (size_t)lo * Dk;
  const bf16x16 qa0 = ld_frag(qrow + hi * 16,      qrow + hi * 16 + 8);
  const bf16x16 qa1 = ld_frag(qrow + 32 + hi * 16, qrow + 32 + hi * 16 + 8);

  f32x8 oacc[4] = {};            // 16x64 output accumulator (4 N-tiles)
  float mrow[8], lrow[8];
#pragma unroll
  for (int r = 0; r < 8; ++r) { mrow[r] = -1e30f; lrow[r] = 0.f; }

  __shared__ __align__(16) unsigned short pT[16][40];  // 16x32 P + pad

  const int qmax = q0 + 15;
  for (int kb = 0; kb <= qmax; kb += 32) {
    // ---- scores: S = Q * K^T for two 16-wide key tiles ----
    f32x8 sc0 = {}, sc1 = {};
    {
      const unsigned short* k0r = kbase + (size_t)(kb + lo) * Dk;
      const unsigned short* k1r = kbase + (size_t)(kb + 16 + lo) * Dk;
      bf16x16 b00 = ld_frag(k0r + hi * 16,      k0r + hi * 16 + 8);
      bf16x16 b01 = ld_frag(k0r + 32 + hi * 16, k0r + 32 + hi * 16 + 8);
      bf16x16 b10 = ld_frag(k1r + hi * 16,      k1r + hi * 16 + 8);
      bf16x16 b11 = ld_frag(k1r + 32 + hi * 16, k1r + 32 + hi * 16 + 8);
      sc0 = wmma_bf16(qa0, b00, sc0);
      sc0 = wmma_bf16(qa1, b01, sc0);
      sc1 = wmma_bf16(qa0, b10, sc1);
      sc1 = wmma_bf16(qa1, b11, sc1);
    }
    if (kb + 32 <= qmax)  // hint next key block toward the caches
      __builtin_prefetch(kbase + (size_t)(kb + 32 + lo) * Dk, 0, 1);

    // ---- online softmax (rows r + 8*hi; columns across 16 lanes) ----
#pragma unroll
    for (int r = 0; r < 8; ++r) {
      const int qpos = q0 + r + hi * 8;
      const int k0 = kb + lo, k1 = kb + 16 + lo;
      float s0 = (k0 <= qpos) ? sc0[r] * 0.125f : -1e30f;
      float s1 = (k1 <= qpos) ? sc1[r] * 0.125f : -1e30f;
      float mnew = fmaxf(mrow[r], rowmax16(fmaxf(s0, s1)));
      float corr = __expf(mrow[r] - mnew);
      float p0 = __expf(s0 - mnew);
      float p1 = __expf(s1 - mnew);
      lrow[r] = lrow[r] * corr + rowsum16(p0 + p1);
      mrow[r] = mnew;
#pragma unroll
      for (int dt = 0; dt < 4; ++dt) oacc[dt][r] *= corr;
      pT[r + hi * 8][lo]      = f2bf(p0);   // C-layout -> row-major in LDS
      pT[r + hi * 8][16 + lo] = f2bf(p1);
    }
    asm volatile("s_wait_dscnt 0" ::: "memory");  // stores visible to reads

    // ---- O += P * V : reload P in A-layout, V^T gives contiguous B rows ----
    bf16x16 pa = ld_frag(&pT[lo][hi * 16], &pT[lo][hi * 16 + 8]);
#pragma unroll
    for (int dt = 0; dt < 4; ++dt) {
      const unsigned short* vrow =
          vbase + (size_t)(dt * 16 + lo) * Sn + kb + hi * 16;
      bf16x16 vb = ld_frag(vrow, vrow + 8);
      oacc[dt] = wmma_bf16(pa, vb, oacc[dt]);
    }
    asm volatile("s_wait_dscnt 0" ::: "memory");  // reads done before rewrite
  }

  // ---- normalize and store head-concatenated O as bf16 [B,S,Dm] ----
#pragma unroll
  for (int r = 0; r < 8; ++r) {
    const float inv = 1.f / lrow[r];
    const int s = q0 + r + hi * 8;
    unsigned short* orow = Ob + ((size_t)b * Sn + s) * Dm + h * Dk;
#pragma unroll
    for (int dt = 0; dt < 4; ++dt)
      orow[dt * 16 + lo] = f2bf(oacc[dt][r] * inv);
  }
}

// ---------------------------------------------------------------------------
// Kernel 3: output projection out = O * Wo^T (fp32 result), 16x64 per wave
// ---------------------------------------------------------------------------
__global__ __launch_bounds__(128)
void out_proj_gemm(const unsigned short* __restrict__ Ob,
                   const unsigned short* __restrict__ wob,
                   float* __restrict__ out) {
  const int lane  = threadIdx.x & 31;
  const int wave  = threadIdx.x >> 5;
  const int hi    = lane >> 4;
  const int lo    = lane & 15;
  const int mbase = blockIdx.x * 16;
  const int ebase = (blockIdx.y * 4 + wave) * 64;

  const unsigned short* arow  = Ob  + (size_t)(mbase + lo) * Dm;
  const unsigned short* brow0 = wob + (size_t)(ebase + lo) * Dm;
  const unsigned short* brow1 = brow0 + (size_t)16 * Dm;
  const unsigned short* brow2 = brow0 + (size_t)32 * Dm;
  const unsigned short* brow3 = brow0 + (size_t)48 * Dm;

  f32x8 acc[4] = {};
  for (int kb = 0; kb < Dm; kb += 32) {
    bf16x16 a = ld_frag(arow + kb + hi * 16, arow + kb + hi * 16 + 8);
    bf16x16 b0 = ld_frag(brow0 + kb + hi * 16, brow0 + kb + hi * 16 + 8);
    bf16x16 b1 = ld_frag(brow1 + kb + hi * 16, brow1 + kb + hi * 16 + 8);
    bf16x16 b2 = ld_frag(brow2 + kb + hi * 16, brow2 + kb + hi * 16 + 8);
    bf16x16 b3 = ld_frag(brow3 + kb + hi * 16, brow3 + kb + hi * 16 + 8);
    acc[0] = wmma_bf16(a, b0, acc[0]);
    acc[1] = wmma_bf16(a, b1, acc[1]);
    acc[2] = wmma_bf16(a, b2, acc[2]);
    acc[3] = wmma_bf16(a, b3, acc[3]);
  }
#pragma unroll
  for (int dt = 0; dt < 4; ++dt) {
#pragma unroll
    for (int r = 0; r < 8; ++r) {
      const int m = mbase + r + hi * 8;
      out[(size_t)m * Dm + ebase + dt * 16 + lo] = acc[dt][r];
    }
  }
}

// ---------------------------------------------------------------------------
extern "C" void kernel_launch(void* const* d_in, const int* in_sizes, int n_in,
                              void* d_out, int out_size, void* d_ws,
                              size_t ws_size, hipStream_t stream) {
  (void)in_sizes; (void)n_in; (void)out_size; (void)ws_size;

  const float* x  = (const float*)d_in[0];
  const int*   tk = (const int*)d_in[1];
  const float* Wq = (const float*)d_in[2];
  const float* Wk = (const float*)d_in[3];
  const float* Wv = (const float*)d_in[4];
  const float* Wo = (const float*)d_in[5];
  float* out = (float*)d_out;

  char*  w   = (char*)d_ws;
  size_t off = 0;
  auto alloc = [&](size_t bytes) -> void* {
    void* p = (void*)(w + off);
    off += (bytes + 255) & ~(size_t)255;
    return p;
  };
  unsigned short* xb  = (unsigned short*)alloc((size_t)Rows * Dm * 2);
  unsigned short* wqb = (unsigned short*)alloc((size_t)Dm * Dm * 2);
  unsigned short* wkb = (unsigned short*)alloc((size_t)Dm * Dm * 2);
  unsigned short* wvb = (unsigned short*)alloc((size_t)Dm * Dm * 2);
  unsigned short* wob = (unsigned short*)alloc((size_t)Dm * Dm * 2);
  unsigned short* Qb  = (unsigned short*)alloc((size_t)Rows * Dm * 2);
  unsigned short* Kb  = (unsigned short*)alloc((size_t)Rows * Dm * 2);
  unsigned short* VT  = (unsigned short*)alloc((size_t)Rows * Dm * 2);
  unsigned short* Ob  = (unsigned short*)alloc((size_t)Rows * Dm * 2);

  // 0) fp32 -> bf16
  cvt_f32_bf16<<<4096, 256, 0, stream>>>(x, xb, Rows * Dm);
  cvt_f32_bf16<<<2048, 256, 0, stream>>>(Wq, wqb, Dm * Dm);
  cvt_f32_bf16<<<2048, 256, 0, stream>>>(Wk, wkb, Dm * Dm);
  cvt_f32_bf16<<<2048, 256, 0, stream>>>(Wv, wvb, Dm * Dm);
  cvt_f32_bf16<<<2048, 256, 0, stream>>>(Wo, wob, Dm * Dm);

  // 1) QKV projection + RoPE (z: 0=Q, 1=K, 2=V); 16x64 tile per wave
  qkv_rope_gemm<<<dim3(Rows / 16, Dm / 256, 3), 128, 0, stream>>>(
      xb, wqb, wkb, wvb, tk, Qb, Kb, VT);

  // 2) causal flash attention, one wave per 16 query rows per (b, h)
  flash_attn<<<dim3(Sn / 16, Bn * Hn), 32, 0, stream>>>(Qb, Kb, VT, Ob);

  // 3) output projection; 16x64 tile per wave
  out_proj_gemm<<<dim3(Rows / 16, Dm / 256), 128, 0, stream>>>(Ob, wob, out);
}
